// MAXViTBlock_12704513262131
// MI455X (gfx1250) — matrix operand order repaired and checked by reading, not compiled, not run on hardware
//
#include <hip/hip_runtime.h>
#include <math.h>

typedef _Float16 h16;
typedef __attribute__((ext_vector_type(8)))  _Float16 v8h;
typedef __attribute__((ext_vector_type(16))) _Float16 v16h;
typedef __attribute__((ext_vector_type(8)))  float    v8f;
typedef __attribute__((ext_vector_type(4)))  unsigned v4u;
typedef __attribute__((ext_vector_type(8)))  unsigned v8u;

#define NTOK 100352          // B*H*W = 32*56*56
#define HWSZ 3136            // 56*56
#define CCH  256
#define BLS_STRIDE 272       // halves: 512B data + 32B pad per row
#define BLS_BUF (64 * BLS_STRIDE)   // halves per LDS weight buffer

// ---------------------------------------------------------------------------
// helpers
// ---------------------------------------------------------------------------
__device__ __forceinline__ v16h frag_from(const h16* lo_p, const h16* hi_p) {
  v8h lo = *(const v8h*)lo_p;
  v8h hi = *(const v8h*)hi_p;
  v16h r;
#pragma unroll
  for (int i = 0; i < 8; ++i) { r[i] = lo[i]; r[i + 8] = hi[i]; }
  return r;
}

__device__ __forceinline__ v8f wmma_f16(v16h a, v16h b, v8f c) {
  // D = A(16x32 f16) * B(32x16 f16) + C(16x16 f32)
  return __builtin_amdgcn_wmma_f32_16x16x32_f16(false, a, false, b, (short)0, c,
                                                false, false);
}

// TDM: DMA a 64-row x 512-byte 2D weight tile (row stride K halves) into LDS,
// with 32B LDS padding after each 512B row (bank-conflict-free ds reads).
__device__ __forceinline__ void tdm_load_tile(unsigned lds_off, const h16* gptr,
                                              int K) {
  const unsigned long long ga = (unsigned long long)(uintptr_t)gptr;
  v4u g0;
  g0[0] = 1u;                                   // count=1, no gather
  g0[1] = lds_off;                              // LDS byte address
  g0[2] = (unsigned)ga;                         // global_addr[31:0]
  g0[3] = (unsigned)(ga >> 32) | 0x80000000u;   // global_addr[56:32] | type=2
  v8u g1;
  g1[0] = (3u << 16)      // data_size = 8 bytes
        | (1u << 20)      // pad_enable
        | (6u << 22)      // pad_interval: every 128 DWORDs (512B)
        | (7u << 25);     // pad_amount: 8 DWORDs (32B)
  g1[1] = (64u << 16);    // tensor_dim0 = 64 units (512B row)
  g1[2] = (64u << 16);    // tensor_dim1 = 64 rows
  g1[3] = (64u << 16);    // tile_dim0 = 64
  g1[4] = 64u;            // tile_dim1 = 64, tile_dim2 = 0
  g1[5] = (unsigned)(K / 4);  // tensor_dim0_stride (8B units)
  g1[6] = 0u;
  g1[7] = 0u;
  asm volatile("tensor_load_to_lds %0, %1" :: "s"(g0), "s"(g1) : "memory");
}

// ---------------------------------------------------------------------------
// Stage 1: depthwise 3x3 conv + bias + LayerNorm  ->  Y(f32, BHWC) + X16(f16)
// one wave per token; lane holds channels c = lane + 32*i
// ---------------------------------------------------------------------------
__global__ __launch_bounds__(256) void conv_ln_kernel(
    const float* __restrict__ x, const float* __restrict__ wdw,
    const float* __restrict__ bdw, const float* __restrict__ g,
    const float* __restrict__ bn, float* __restrict__ Y,
    h16* __restrict__ X16) {
  const int wave = threadIdx.x >> 5, lane = threadIdx.x & 31;
  const int t = blockIdx.x * 8 + wave;
  const int b = t / HWSZ, hw = t - b * HWSZ;
  const int h = hw / 56, w = hw - h * 56;

  float vals[8];
#pragma unroll
  for (int i = 0; i < 8; ++i) {
    const int c = lane + 32 * i;
    const float* xp = x + ((size_t)(b * CCH + c)) * HWSZ;
    const float* wp = wdw + c * 9;
    float acc = bdw[c];
#pragma unroll
    for (int dy = -1; dy <= 1; ++dy) {
      const int hh = h + dy;
      if (hh < 0 || hh >= 56) continue;
#pragma unroll
      for (int dx = -1; dx <= 1; ++dx) {
        const int ww = w + dx;
        if (ww < 0 || ww >= 56) continue;
        acc += xp[hh * 56 + ww] * wp[(dy + 1) * 3 + (dx + 1)];
      }
    }
    vals[i] = acc;
  }
  float s = 0.f;
#pragma unroll
  for (int i = 0; i < 8; ++i) s += vals[i];
  for (int off = 1; off < 32; off <<= 1) s += __shfl_xor(s, off, 32);
  const float mu = s * (1.f / 256.f);
  float vs = 0.f;
#pragma unroll
  for (int i = 0; i < 8; ++i) { float d = vals[i] - mu; vs += d * d; }
  for (int off = 1; off < 32; off <<= 1) vs += __shfl_xor(vs, off, 32);
  const float inv = rsqrtf(vs * (1.f / 256.f) + 1e-5f);
#pragma unroll
  for (int i = 0; i < 8; ++i) {
    const int c = lane + 32 * i;
    const float yv = (vals[i] - mu) * inv * g[c] + bn[c];
    const size_t idx = (size_t)t * CCH + c;
    Y[idx] = yv;
    X16[idx] = (h16)yv;
  }
}

// LayerNorm only: Y -> X16 (f16), Y unchanged
__global__ __launch_bounds__(256) void ln_kernel(const float* __restrict__ Y,
                                                 const float* __restrict__ g,
                                                 const float* __restrict__ bn,
                                                 h16* __restrict__ X16) {
  const int wave = threadIdx.x >> 5, lane = threadIdx.x & 31;
  const int t = blockIdx.x * 8 + wave;
  float vals[8];
#pragma unroll
  for (int i = 0; i < 8; ++i) vals[i] = Y[(size_t)t * CCH + lane + 32 * i];
  float s = 0.f;
#pragma unroll
  for (int i = 0; i < 8; ++i) s += vals[i];
  for (int off = 1; off < 32; off <<= 1) s += __shfl_xor(s, off, 32);
  const float mu = s * (1.f / 256.f);
  float vs = 0.f;
#pragma unroll
  for (int i = 0; i < 8; ++i) { float d = vals[i] - mu; vs += d * d; }
  for (int off = 1; off < 32; off <<= 1) vs += __shfl_xor(vs, off, 32);
  const float inv = rsqrtf(vs * (1.f / 256.f) + 1e-5f);
#pragma unroll
  for (int i = 0; i < 8; ++i) {
    const int c = lane + 32 * i;
    X16[(size_t)t * CCH + c] = (h16)((vals[i] - mu) * inv * g[c] + bn[c]);
  }
}

// f32 -> f16 weight conversion
__global__ void cvt_f2h(const float* __restrict__ in, h16* __restrict__ out, int n) {
  const int i = blockIdx.x * 256 + threadIdx.x;
  if (i < n) out[i] = (h16)in[i];
}

// ---------------------------------------------------------------------------
// WMMA NT GEMM:  C[M,N] = A[M,K](f16, row-major) * W[N,K]^T(f16) + bias
// 256 threads = 8 waves; wave computes 32(M) x 64(N); block tile 256 x 64.
// Weight tiles (64 x 256 f16) staged into LDS by the Tensor Data Mover with
// double-buffered chunks over K; A fragments register-double-buffered from
// global. Per k-step: 4 B frags in distinct regs -> one ds wait, 8 WMMAs.
// EP: 0 = store f16 ; 1 = GELU -> f16 ; 2 = Y += v, X16 = f16(Y)
//     3 = final: out(NCHW) = xin + Y + v
// ---------------------------------------------------------------------------
template <int EP>
__global__ __launch_bounds__(256) void gemm_nt(
    const h16* __restrict__ A, const h16* __restrict__ Bw,
    const float* __restrict__ bias, int M, int N, int K,
    h16* __restrict__ outh, float* __restrict__ Yacc, h16* __restrict__ x16,
    const float* __restrict__ xin, float* __restrict__ outf) {
  __shared__ alignas(64) h16 Bls[2 * BLS_BUF];

  const int wave = threadIdx.x >> 5, lane = threadIdx.x & 31;
  const int row0 = blockIdx.y * 256 + wave * 32;
  const int col0 = blockIdx.x * 64;
  const unsigned bls_off = (unsigned)(uintptr_t)(void*)&Bls[0];

  v8f z = {};
  v8f acc[8];
#pragma unroll
  for (int t = 0; t < 8; ++t) acc[t] = z;

  const int ra = lane & 15, ka = (lane >> 4) * 8;    // A-frag lane mapping
  const int nb = lane & 15, kbb = (lane >> 4) * 8;  // B-frag lane mapping
  const h16* ap0 = A + (size_t)(row0 + ra) * K + ka;
  const h16* ap1 = ap0 + (size_t)16 * K;

  // prefetch first A fragments
  v16h a0n = frag_from(ap0, ap0 + 16);
  v16h a1n = frag_from(ap1, ap1 + 16);

  const int nch = K >> 8;
  if (threadIdx.x < 32) tdm_load_tile(bls_off, Bw + (size_t)col0 * K, K);

  for (int c = 0; c < nch; ++c) {
    if (threadIdx.x < 32) {
      if (c + 1 < nch) {
        // prefetch next chunk into the other buffer, then wait current done
        tdm_load_tile(bls_off + (unsigned)(((c + 1) & 1) * BLS_BUF * 2),
                      Bw + (size_t)col0 * K + (c + 1) * 256, K);
        __builtin_amdgcn_s_wait_tensorcnt(1);
      } else {
        __builtin_amdgcn_s_wait_tensorcnt(0);
      }
    }
    __syncthreads();
    const h16* bbase = &Bls[(c & 1) * BLS_BUF];

#pragma unroll
    for (int k0 = 0; k0 < 256; k0 += 32) {
      v16h a0 = a0n, a1 = a1n;
      int knext = c * 256 + k0 + 32;
      if (knext >= K) knext = 0;  // harmless tail re-read
      a0n = frag_from(ap0 + knext, ap0 + knext + 16);
      a1n = frag_from(ap1 + knext, ap1 + knext + 16);

      const h16* bp0 = bbase + (0 * 16 + nb) * BLS_STRIDE + k0 + kbb;
      const h16* bp1 = bbase + (1 * 16 + nb) * BLS_STRIDE + k0 + kbb;
      const h16* bp2 = bbase + (2 * 16 + nb) * BLS_STRIDE + k0 + kbb;
      const h16* bp3 = bbase + (3 * 16 + nb) * BLS_STRIDE + k0 + kbb;
      v16h b0 = frag_from(bp0, bp0 + 16);
      v16h b1 = frag_from(bp1, bp1 + 16);
      v16h b2 = frag_from(bp2, bp2 + 16);
      v16h b3 = frag_from(bp3, bp3 + 16);

      acc[0] = wmma_f16(a0, b0, acc[0]);
      acc[1] = wmma_f16(a0, b1, acc[1]);
      acc[2] = wmma_f16(a0, b2, acc[2]);
      acc[3] = wmma_f16(a0, b3, acc[3]);
      acc[4] = wmma_f16(a1, b0, acc[4]);
      acc[5] = wmma_f16(a1, b1, acc[5]);
      acc[6] = wmma_f16(a1, b2, acc[6]);
      acc[7] = wmma_f16(a1, b3, acc[7]);
    }
    __syncthreads();  // waves done reading this buffer before it is re-filled
  }

#pragma unroll
  for (int ms = 0; ms < 2; ++ms) {
    const int mb = row0 + ms * 16 + ((lane >> 4) * 8);
#pragma unroll
    for (int t = 0; t < 4; ++t) {
      const int col = col0 + t * 16 + (lane & 15);
      const float bv = bias[col];
#pragma unroll
      for (int i = 0; i < 8; ++i) {
        const int row = mb + i;
        float v = acc[ms * 4 + t][i] + bv;
        if (EP == 0) {
          outh[(size_t)row * N + col] = (h16)v;
        } else if (EP == 1) {
          const float ge = 0.5f * v * (1.f + erff(v * 0.70710678118654752f));
          outh[(size_t)row * N + col] = (h16)ge;
        } else if (EP == 2) {
          const size_t idx = (size_t)row * N + col;
          const float nv = Yacc[idx] + v;
          Yacc[idx] = nv;
          x16[idx] = (h16)nv;
        } else {  // EP == 3: final residual + NHWC->NCHW transpose
          const size_t idx = (size_t)row * CCH + col;
          const int b = row / HWSZ, hw = row - b * HWSZ;
          const size_t oi = ((size_t)(b * CCH + col)) * HWSZ + hw;
          outf[oi] = xin[oi] + Yacc[idx] + v;
        }
      }
    }
  }
}

// ---------------------------------------------------------------------------
// Attention: one block = one (sequence, head); 4 waves; L <= 64.
// qkv layout: [token, 768] f16, q|k|v segments of 256, head dim 32.
// ---------------------------------------------------------------------------
template <int L, bool WINDOW>
__global__ __launch_bounds__(128) void attn_kernel(const h16* __restrict__ qkv,
                                                   h16* __restrict__ o16) {
  alignas(16) __shared__ h16 Pls[4][16][72];
  alignas(16) __shared__ h16 Vt[32][72];

  const int seq = blockIdx.x, head = blockIdx.y;
  const int tid = threadIdx.x, wave = tid >> 5, lane = tid & 31;
  const float scale = 0.17677669529663687f;  // 1/sqrt(32)

  const h16* qbase = qkv + head * 32;
  const h16* kbase = qkv + 256 + head * 32;
  const h16* vbase = qkv + 512 + head * 32;

  auto tok = [&](int pos) -> size_t {
    pos = min(pos, L - 1);
    if (!WINDOW) return (size_t)seq * 56 + pos;
    const int b = seq >> 6, wy = (seq >> 3) & 7, wx = seq & 7;
    const int iy = pos / 7, ix = pos - iy * 7;
    return (size_t)((b * 56 + wy * 7 + iy) * 56 + wx * 7 + ix);
  };

  // stage V^T [d][j] into LDS (zero-padded beyond L)
  for (int idx = tid; idx < 32 * 64; idx += 128) {
    const int d = idx >> 6, j = idx & 63;
    Vt[d][j] = (j < L) ? vbase[tok(j) * 768 + d] : (h16)0.f;
  }
  __syncthreads();

  // Q fragment: wave handles q rows [16*wave, 16*wave+16)
  v16h qa;
  {
    const int r = lane & 15, kq = (lane >> 4) * 8;
    const h16* p = qbase + tok(wave * 16 + r) * 768 + kq;
    qa = frag_from(p, p + 16);
  }

  // S = Q K^T : 4 n-tiles of 16 keys (64 >= L, masked later)
  v8f z = {};
  v8f s[4];
#pragma unroll
  for (int t = 0; t < 4; ++t) {
    const int n = lane & 15, kk = (lane >> 4) * 8;
    const h16* p = kbase + tok(t * 16 + n) * 768 + kk;
    v16h kf = frag_from(p, p + 16);
    s[t] = wmma_f16(qa, kf, z);
  }

  // row-wise softmax; row M = i + (lane>>4)*8, col j = t*16 + (lane&15)
  const int jcol = lane & 15;
#pragma unroll
  for (int i = 0; i < 8; ++i) {
    float mx = -1e30f;
#pragma unroll
    for (int t = 0; t < 4; ++t) {
      const float sv = (t * 16 + jcol < L) ? s[t][i] * scale : -1e30f;
      s[t][i] = sv;
      mx = fmaxf(mx, sv);
    }
    for (int off = 1; off < 16; off <<= 1) mx = fmaxf(mx, __shfl_xor(mx, off, 32));
    float sm = 0.f;
#pragma unroll
    for (int t = 0; t < 4; ++t) {
      float e = __expf(s[t][i] - mx);
      if (t * 16 + jcol >= L) e = 0.f;
      s[t][i] = e;
      sm += e;
    }
    for (int off = 1; off < 16; off <<= 1) sm += __shfl_xor(sm, off, 32);
    const float inv = 1.f / sm;
    const int row = i + ((lane >> 4) * 8);
#pragma unroll
    for (int t = 0; t < 4; ++t)
      Pls[wave][row][t * 16 + jcol] = (h16)(s[t][i] * inv);
  }
  __syncthreads();

  // O = P(16x64) * V(64x32) : 2 k-steps x 2 n-tiles
  v8f oacc[2];
  oacc[0] = z; oacc[1] = z;
#pragma unroll
  for (int kt = 0; kt < 2; ++kt) {
    const int r = lane & 15, kp = (lane >> 4) * 8;
    const h16* pp = &Pls[wave][r][kt * 32 + kp];
    v16h pa = frag_from(pp, pp + 16);
#pragma unroll
    for (int nt = 0; nt < 2; ++nt) {
      const int d = nt * 16 + (lane & 15);
      const h16* vp = &Vt[d][kt * 32 + (lane >> 4) * 8];
      v16h vb = frag_from(vp, vp + 16);
      oacc[nt] = wmma_f16(pa, vb, oacc[nt]);
    }
  }

  // store O rows < L (f16, token order, heads concatenated)
  const int mbase = (lane >> 4) * 8;
#pragma unroll
  for (int nt = 0; nt < 2; ++nt) {
    const int d = nt * 16 + (lane & 15);
#pragma unroll
    for (int i = 0; i < 8; ++i) {
      const int qrow = wave * 16 + mbase + i;
      if (qrow < L) o16[tok(qrow) * CCH + head * 32 + d] = (h16)oacc[nt][i];
    }
  }
}

// ---------------------------------------------------------------------------
// launcher
// ---------------------------------------------------------------------------
extern "C" void kernel_launch(void* const* d_in, const int* in_sizes, int n_in,
                              void* d_out, int out_size, void* d_ws,
                              size_t ws_size, hipStream_t stream) {
  (void)in_sizes; (void)n_in; (void)out_size; (void)ws_size;
  const float* x      = (const float*)d_in[0];
  const float* dw_w   = (const float*)d_in[1];
  const float* dw_b   = (const float*)d_in[2];
  const float* g1     = (const float*)d_in[3];
  const float* b1n    = (const float*)d_in[4];
  const float* Wqkv_g = (const float*)d_in[5];
  const float* bqkv_g = (const float*)d_in[6];
  const float* Wo_g   = (const float*)d_in[7];
  const float* bo_g   = (const float*)d_in[8];
  const float* Wqkv_w = (const float*)d_in[9];
  const float* bqkv_w = (const float*)d_in[10];
  const float* Wo_w   = (const float*)d_in[11];
  const float* bo_w   = (const float*)d_in[12];
  const float* g3     = (const float*)d_in[13];
  const float* b3n    = (const float*)d_in[14];
  const float* W1     = (const float*)d_in[15];
  const float* b1     = (const float*)d_in[16];
  const float* W2     = (const float*)d_in[17];
  const float* b2     = (const float*)d_in[18];
  float* out = (float*)d_out;

  char* ws = (char*)d_ws;
  size_t off = 0;
  float* Y  = (float*)(ws + off); off += (size_t)NTOK * CCH * 4;
  h16* X16  = (h16*)(ws + off);   off += (size_t)NTOK * CCH * 2;
  h16* BIG  = (h16*)(ws + off);   off += (size_t)NTOK * 1024 * 2;  // qkv / mlp hidden
  h16* O16  = (h16*)(ws + off);   off += (size_t)NTOK * CCH * 2;
  h16* WH   = (h16*)(ws + off);
  h16* WqkvGh = WH;
  h16* WoGh   = WqkvGh + 768 * 256;
  h16* WqkvWh = WoGh + 256 * 256;
  h16* WoWh   = WqkvWh + 768 * 256;
  h16* W1h    = WoWh + 256 * 256;
  h16* W2h    = W1h + 1024 * 256;

  // weights f32 -> f16
  cvt_f2h<<<(768 * 256 + 255) / 256, 256, 0, stream>>>(Wqkv_g, WqkvGh, 768 * 256);
  cvt_f2h<<<(256 * 256 + 255) / 256, 256, 0, stream>>>(Wo_g, WoGh, 256 * 256);
  cvt_f2h<<<(768 * 256 + 255) / 256, 256, 0, stream>>>(Wqkv_w, WqkvWh, 768 * 256);
  cvt_f2h<<<(256 * 256 + 255) / 256, 256, 0, stream>>>(Wo_w, WoWh, 256 * 256);
  cvt_f2h<<<(1024 * 256 + 255) / 256, 256, 0, stream>>>(W1, W1h, 1024 * 256);
  cvt_f2h<<<(1024 * 256 + 255) / 256, 256, 0, stream>>>(W2, W2h, 1024 * 256);

  // 1. depthwise conv + LN -> Y, X16
  conv_ln_kernel<<<NTOK / 8, 256, 0, stream>>>(x, dw_w, dw_b, g1, b1n, Y, X16);

  // 2. grid attention: qkv, per-row attention, projection + residual
  gemm_nt<0><<<dim3(768 / 64, NTOK / 256), 256, 0, stream>>>(
      X16, WqkvGh, bqkv_g, NTOK, 768, 256, BIG, nullptr, nullptr, nullptr, nullptr);
  attn_kernel<56, false><<<dim3(32 * 56, 8), 128, 0, stream>>>(BIG, O16);
  gemm_nt<2><<<dim3(256 / 64, NTOK / 256), 256, 0, stream>>>(
      O16, WoGh, bo_g, NTOK, 256, 256, nullptr, Y, X16, nullptr, nullptr);

  // 3. window attention (7x7), projection + residual
  gemm_nt<0><<<dim3(768 / 64, NTOK / 256), 256, 0, stream>>>(
      X16, WqkvWh, bqkv_w, NTOK, 768, 256, BIG, nullptr, nullptr, nullptr, nullptr);
  attn_kernel<49, true><<<dim3(32 * 64, 8), 128, 0, stream>>>(BIG, O16);
  gemm_nt<2><<<dim3(256 / 64, NTOK / 256), 256, 0, stream>>>(
      O16, WoWh, bo_w, NTOK, 256, 256, nullptr, Y, X16, nullptr, nullptr);

  // 4. MLP: LN -> W1+GELU -> W2 (+Y, +shortcut, NCHW transpose)
  ln_kernel<<<NTOK / 8, 256, 0, stream>>>(Y, g3, b3n, X16);
  gemm_nt<1><<<dim3(1024 / 64, NTOK / 256), 256, 0, stream>>>(
      X16, W1h, b1, NTOK, 1024, 256, BIG, nullptr, nullptr, nullptr, nullptr);
  gemm_nt<3><<<dim3(256 / 64, NTOK / 256), 256, 0, stream>>>(
      BIG, W2h, b2, NTOK, 256, 1024, nullptr, Y, nullptr, x, out);
}
